// AttentionBlock_45157286150731
// MI455X (gfx1250) — matrix operand order repaired and checked by reading, not compiled, not run on hardware
//
#include <hip/hip_runtime.h>

typedef __attribute__((ext_vector_type(16))) _Float16 v16h;
typedef __attribute__((ext_vector_type(8)))  float    v8f;
typedef __attribute__((ext_vector_type(4)))  float    v4f;

#define NB   4
#define NC   64
#define HWPX 4096
#define QKD  8

// ---------------------------------------------------------------------------
// Kernel 1: fused 1x1-conv projections.
// One thread per (b, pixel). Reads the 64-channel column of x once, produces
// q[8], k[8] (fp32, pixel-major for the score loop) and v[64] (f16, channel-
// major rows so the flash kernel can load WMMA B operands contiguously).
// q is pre-scaled by log2(e) so the flash kernel's softmax runs in exp2 domain
// (v_exp_f32 is natively exp2 on AMD -> saves a v_mul per exponential).
// ---------------------------------------------------------------------------
__global__ __launch_bounds__(256) void attn_proj_kernel(
    const float* __restrict__ x,
    const float* __restrict__ Wq, const float* __restrict__ bq,
    const float* __restrict__ Wk, const float* __restrict__ bk,
    const float* __restrict__ Wv, const float* __restrict__ bv,
    float* __restrict__ qf, float* __restrict__ kf, _Float16* __restrict__ vh)
{
    const float LOG2E = 1.44269504088896340736f;
    const int t = blockIdx.x * blockDim.x + threadIdx.x;   // (b, p) flat
    const int b = t >> 12;
    const int p = t & (HWPX - 1);

    const float* xp = x + (size_t)b * NC * HWPX + p;
    float xc[NC];
#pragma unroll
    for (int c = 0; c < NC; ++c) xc[c] = xp[(size_t)c * HWPX];

    // q and k projections (8 outputs each)
#pragma unroll
    for (int d = 0; d < QKD; ++d) {
        float aq = bq[d], ak = bk[d];
        const float* wq = Wq + d * NC;
        const float* wk = Wk + d * NC;
#pragma unroll
        for (int c = 0; c < NC; ++c) {
            aq = fmaf(wq[c], xc[c], aq);
            ak = fmaf(wk[c], xc[c], ak);
        }
        qf[(size_t)t * QKD + d] = aq * LOG2E;   // exp2-domain scores
        kf[(size_t)t * QKD + d] = ak;
        }

    // v projection (64 outputs, f16, stored [b][c][p]; coalesced across lanes)
    for (int c = 0; c < NC; ++c) {
        float av = bv[c];
        const float* wv = Wv + c * NC;
#pragma unroll
        for (int cc = 0; cc < NC; ++cc) av = fmaf(wv[cc], xc[cc], av);
        vh[((size_t)b * NC + c) * HWPX + p] = (_Float16)av;
    }
}

// ---------------------------------------------------------------------------
// Kernel 2: flash attention. One wave32 owns a 16-query tile; streams j in
// chunks of 64 (two 16x32 A tiles -> 8 WMMAs per iteration). Scores: fp32
// VALU (d=8), computed directly in the contiguous k order this kernel uses for the WMMA f16
// A-operand (lane<16: K∈{0..15} of its row, lane>=16: K∈{16..31}).
// Online softmax state (m, l) for row i lives in lane pair (i, i+16); the O
// accumulator rescale is skipped entirely (uniform s_cbranch) on iterations
// where no row's running max changed.
// ---------------------------------------------------------------------------
__global__ __launch_bounds__(128) void attn_flash_kernel(
    const float* __restrict__ qf, const float* __restrict__ kf,
    const _Float16* __restrict__ vh, const float* __restrict__ x,
    const float* __restrict__ gamma, float* __restrict__ out)
{
    const int lane = threadIdx.x & 31;
    const int wave = threadIdx.x >> 5;
    const int tile = blockIdx.x * 4 + wave;       // 1024 tiles total
    const int b    = tile >> 8;                   // 256 tiles per batch
    const int ib   = (tile & 255) << 4;           // query-row base
    const bool hi  = lane >= 16;
    const int  l16 = lane & 15;
    const int  myI = ib + l16;                    // softmax row owned by lane pair

    // q row for this lane's row (8 floats, already *log2e)
    const v4f* qp = (const v4f*)(qf + ((size_t)b * HWPX + myI) * QKD);
    const v4f q0 = qp[0], q1 = qp[1];

    float m = -1e30f, lsum = 0.0f;
    v8f acc[4];
#pragma unroll
    for (int g = 0; g < 4; ++g)
#pragma unroll
        for (int e = 0; e < 8; ++e) acc[g][e] = 0.0f;

    const _Float16* vbase = vh + (size_t)b * NC * HWPX;
    const float*    kbase = kf + (size_t)b * HWPX * QKD;
    const int rsel = (hi ? 8 : 0);

    for (int jb = 0; jb < HWPX; jb += 64) {
        // ---- scores for 64 j (two A tiles), in WMMA A slot order ----
        float s[32];
#pragma unroll
        for (int t2 = 0; t2 < 2; ++t2) {
#pragma unroll
            for (int h = 0; h < 16; ++h) {
                const int joff = t2 * 32 + h + (hi ? 16 : 0);
                const v4f* kp = (const v4f*)(kbase + (size_t)(jb + joff) * QKD);
                const v4f k0 = kp[0], k1 = kp[1];
                float d0 = q0.x * k0.x + q0.y * k0.y + q0.z * k0.z + q0.w * k0.w;
                float d1 = q1.x * k1.x + q1.y * k1.y + q1.z * k1.z + q1.w * k1.w;
                s[t2 * 16 + h] = d0 + d1;
            }
        }
        // ---- online softmax: row max over lane pair ----
        float mloc = s[0];
#pragma unroll
        for (int h = 1; h < 32; ++h) mloc = fmaxf(mloc, s[h]);
        mloc = fmaxf(mloc, __shfl_xor(mloc, 16, 32));
        const float mnew = fmaxf(m, mloc);

        // rescale O/l only when some row's max actually moved (uniform branch)
        if (__ballot(mloc > m) != 0ull) {
            const float scale = __builtin_amdgcn_exp2f(m - mnew);
            lsum *= scale;
#pragma unroll
            for (int r = 0; r < 8; ++r) {
                const float fr = __shfl(scale, r + rsel, 32);
#pragma unroll
                for (int g = 0; g < 4; ++g) acc[g][r] *= fr;
            }
        }
        m = mnew;

        v16h a0, a1;
        float ls = 0.0f;
#pragma unroll
        for (int h = 0; h < 16; ++h) {
            const float p = __builtin_amdgcn_exp2f(s[h] - mnew);
            ls += p;
            a0[h] = (_Float16)p;
        }
#pragma unroll
        for (int h = 0; h < 16; ++h) {
            const float p = __builtin_amdgcn_exp2f(s[16 + h] - mnew);
            ls += p;
            a1[h] = (_Float16)p;
        }
        lsum += ls;

        // ---- PV: 8 WMMAs cover c = 0..63 for 64 j ----
#pragma unroll
        for (int g = 0; g < 4; ++g) {
            const _Float16* vp =
                vbase + (size_t)(g * 16 + l16) * HWPX + jb + (hi ? 16 : 0);
            const v16h bm0 = *(const v16h*)vp;         // j chunk 0: K half
            const v16h bm1 = *(const v16h*)(vp + 32);  // j chunk 1
            acc[g] = __builtin_amdgcn_wmma_f32_16x16x32_f16(
                false, a0, false, bm0, (short)0, acc[g], false, false);
            acc[g] = __builtin_amdgcn_wmma_f32_16x16x32_f16(
                false, a1, false, bm1, (short)0, acc[g], false, false);
        }
    }

    // ---- epilogue: normalize, gamma * O + x ----
    const float ltot = lsum + __shfl_xor(lsum, 16, 32);
    const float gam  = gamma[0];
#pragma unroll
    for (int r = 0; r < 8; ++r) {
        const float lr  = __shfl(ltot, r + rsel, 32);
        const float inv = __builtin_amdgcn_rcpf(lr);
        const int   i   = ib + r + rsel;
#pragma unroll
        for (int g = 0; g < 4; ++g) {
            const int    c   = g * 16 + l16;
            const size_t idx = ((size_t)b * NC + c) * HWPX + i;
            out[idx] = gam * (acc[g][r] * inv) + x[idx];
        }
    }
}

// ---------------------------------------------------------------------------
extern "C" void kernel_launch(void* const* d_in, const int* in_sizes, int n_in,
                              void* d_out, int out_size, void* d_ws, size_t ws_size,
                              hipStream_t stream) {
    const float* x     = (const float*)d_in[0];
    const float* Wq    = (const float*)d_in[1];
    const float* bq    = (const float*)d_in[2];
    const float* Wk    = (const float*)d_in[3];
    const float* bk    = (const float*)d_in[4];
    const float* Wv    = (const float*)d_in[5];
    const float* bv    = (const float*)d_in[6];
    const float* gamma = (const float*)d_in[7];
    float* out = (float*)d_out;

    // workspace: qf (512KB) | kf (512KB) | vh (2MB)
    char* ws = (char*)d_ws;
    const size_t qk_bytes = (size_t)NB * HWPX * QKD * sizeof(float);
    float*    qf = (float*)ws;
    float*    kf = (float*)(ws + qk_bytes);
    _Float16* vh = (_Float16*)(ws + 2 * qk_bytes);

    attn_proj_kernel<<<(NB * HWPX) / 256, 256, 0, stream>>>(
        x, Wq, bq, Wk, bk, Wv, bv, qf, kf, vh);

    // 1024 query tiles (4 batches x 256 tiles), 4 waves per block
    attn_flash_kernel<<<256, 128, 0, stream>>>(qf, kf, vh, x, gamma, out);
}
